// Induction_86844238725131
// MI455X (gfx1250) — matrix operand order repaired and checked by reading, not compiled, not run on hardware
//
#include <hip/hip_runtime.h>

#define C_CAPS 512
#define S_DIM  128
#define H_DIM  1024
#define ITERS  3
#define EPSV   1e-9f

typedef __attribute__((ext_vector_type(16))) _Float16 v16h;
typedef __attribute__((ext_vector_type(8)))  _Float16 v8h;
typedef __attribute__((ext_vector_type(4)))  _Float16 v4h;
typedef __attribute__((ext_vector_type(8)))  float    v8f;

#define E_STRIDE  1032   // halfs per e row (1024 + 8 pad, keeps 16B alignment)
#define WT_STRIDE 40     // halfs per Wt column in LDS-staging fallback

// ---------------------------------------------------------------------------
// Shared phase 2: routing iterations entirely out of LDS-resident e (f16).
// ---------------------------------------------------------------------------
__device__ __forceinline__
void routing_phase(_Float16* eL, float* cvec, float* bij, float* dvec,
                   float* bpart, int tid, int cap, float* __restrict__ out) {
  const int h0 = tid * 4;   // 4 contiguous h per thread
  for (int it = 0; it < ITERS; ++it) {
    // softmax over s (redundant scan, LDS broadcast reads)
    if (tid < S_DIM) {
      float mx = -3.4e38f;
      for (int s = 0; s < S_DIM; ++s) mx = fmaxf(mx, bij[s]);
      float sum = 0.f;
      for (int s = 0; s < S_DIM; ++s) sum += __expf(bij[s] - mx);
      dvec[tid] = __expf(bij[tid] - mx) / sum;
    }
    __syncthreads();

    // c_h = sum_s d_s * e[s][h]
    float c0 = 0.f, c1 = 0.f, c2 = 0.f, c3 = 0.f;
    for (int s = 0; s < S_DIM; ++s) {
      const float ds = dvec[s];
      v4h e4 = *(const v4h*)&eL[(size_t)s * E_STRIDE + h0];
      c0 += ds * (float)e4[0];
      c1 += ds * (float)e4[1];
      c2 += ds * (float)e4[2];
      c3 += ds * (float)e4[3];
    }

    // squash: deterministic tree reduction of sum(c^2)
    bpart[tid] = c0 * c0 + c1 * c1 + c2 * c2 + c3 * c3;
    __syncthreads();
    for (int off = 128; off > 0; off >>= 1) {
      if (tid < off) bpart[tid] += bpart[tid + off];
      __syncthreads();
    }
    const float sq = bpart[0];
    const float coeff = sq / (1.0f + sq) * rsqrtf(sq + EPSV);
    c0 *= coeff; c1 *= coeff; c2 *= coeff; c3 *= coeff;
    *reinterpret_cast<float4*>(&cvec[h0]) = make_float4(c0, c1, c2, c3);
    __syncthreads();

    // b_s += e[s][:] . c[:]   (skip on last iteration)
    if (it < ITERS - 1) {
      const int s    = tid & 127;
      const int half = tid >> 7;
      float accb = 0.f;
      const _Float16* ep = &eL[(size_t)s * E_STRIDE + half * 512];
      const float*    cp = &cvec[half * 512];
      for (int h = 0; h < 512; h += 4) {
        v4h e4 = *(const v4h*)(ep + h);
        accb += (float)e4[0] * cp[h + 0] + (float)e4[1] * cp[h + 1] +
                (float)e4[2] * cp[h + 2] + (float)e4[3] * cp[h + 3];
      }
      bpart[tid] = accb;
      __syncthreads();
      if (tid < S_DIM) bij[tid] += bpart[tid] + bpart[tid + 128];
      __syncthreads();
    }
  }

  float4 r = make_float4(cvec[h0], cvec[h0 + 1], cvec[h0 + 2], cvec[h0 + 3]);
  *reinterpret_cast<float4*>(out + (size_t)cap * H_DIM + h0) = r;
}

// A fragment: 16x32 f16. lane<16: K{0-15}; lane>=16: K{16-31}.
__device__ __forceinline__ v16h load_a_frag(const float* __restrict__ xr) {
  float4 f0 = *reinterpret_cast<const float4*>(xr + 0);
  float4 f1 = *reinterpret_cast<const float4*>(xr + 4);
  float4 f2 = *reinterpret_cast<const float4*>(xr + 8);
  float4 f3 = *reinterpret_cast<const float4*>(xr + 12);
  v16h a;
  a[0]  = (_Float16)f0.x;  a[1]  = (_Float16)f0.y;
  a[2]  = (_Float16)f0.z;  a[3]  = (_Float16)f0.w;
  a[4]  = (_Float16)f1.x;  a[5]  = (_Float16)f1.y;
  a[6]  = (_Float16)f1.z;  a[7]  = (_Float16)f1.w;
  a[8]  = (_Float16)f2.x;  a[9]  = (_Float16)f2.y;
  a[10] = (_Float16)f2.z;  a[11] = (_Float16)f2.w;
  a[12] = (_Float16)f3.x;  a[13] = (_Float16)f3.y;
  a[14] = (_Float16)f3.z;  a[15] = (_Float16)f3.w;
  return a;
}

// ---------------------------------------------------------------------------
// Prep kernel: Wt[n][k] = (f16) W[k][n].  2 MB, stays L2-resident.
// ---------------------------------------------------------------------------
__global__ __launch_bounds__(256)
void transpose_w_f16(const float* __restrict__ W, _Float16* __restrict__ Wt) {
  __shared__ _Float16 tile[64 * 80];           // 64x64 tile, padded stride 80
  const int tid = threadIdx.x;
  const int n0 = (blockIdx.x & 15) * 64;
  const int k0 = (blockIdx.x >> 4) * 64;

  const int c0 = (tid & 15) * 4;               // tile col (n)
  const int r0 = tid >> 4;                     // tile row (k), 0..15
  #pragma unroll
  for (int rb = 0; rb < 64; rb += 16) {
    const int r = r0 + rb;
    float4 f = *reinterpret_cast<const float4*>(W + (size_t)(k0 + r) * H_DIM + n0 + c0);
    tile[(c0 + 0) * 80 + r] = (_Float16)f.x;
    tile[(c0 + 1) * 80 + r] = (_Float16)f.y;
    tile[(c0 + 2) * 80 + r] = (_Float16)f.z;
    tile[(c0 + 3) * 80 + r] = (_Float16)f.w;
  }
  __syncthreads();

  const int ko  = (tid & 7) * 8;               // k offset within tile
  const int rr0 = tid >> 3;                    // n row within tile, 0..31
  #pragma unroll
  for (int rb = 0; rb < 64; rb += 32) {
    const int rr = rr0 + rb;
    v8h v = *reinterpret_cast<const v8h*>(&tile[rr * 80 + ko]);
    *reinterpret_cast<v8h*>(Wt + (size_t)(n0 + rr) * H_DIM + k0 + ko) = v;
  }
}

// ---------------------------------------------------------------------------
// Fast fused kernel: B fragments straight from pre-transposed f16 W (L2).
// K-loop is barrier-free and software-pipelined (double-buffered A and B).
// ---------------------------------------------------------------------------
static constexpr size_t SMEM_FAST =
    (size_t)S_DIM * E_STRIDE * 2 + H_DIM * 4 + S_DIM * 4 + S_DIM * 4 + 256 * 4;

__global__ __launch_bounds__(256, 1)
void caps_fused_fast(const float* __restrict__ x, const _Float16* __restrict__ WtG,
                     float* __restrict__ out) {
  extern __shared__ char smem[];
  _Float16* eL    = (_Float16*)smem;
  float*    cvec  = (float*)(smem + (size_t)S_DIM * E_STRIDE * 2);
  float*    bij   = cvec + H_DIM;
  float*    dvec  = bij + S_DIM;
  float*    bpart = dvec + S_DIM;

  const int tid  = threadIdx.x;
  const int lane = tid & 31;
  const int wv   = tid >> 5;
  const int l16  = lane & 15;
  const int hi   = lane >> 4;
  const int cap  = blockIdx.x;

  const float* xc = x + (size_t)cap * S_DIM * H_DIM;
  if (tid < S_DIM) bij[tid] = 0.0f;

  const int wrow = wv * 16;
  for (int nc = 0; nc < H_DIM / 128; ++nc) {
    const int nbase = nc * 128;
    v8f acc[8];
    #pragma unroll
    for (int j = 0; j < 8; ++j) {
      v8f z = {0.f, 0.f, 0.f, 0.f, 0.f, 0.f, 0.f, 0.f};
      acc[j] = z;
    }

    const float*    xrow  = xc + (size_t)(wrow + l16) * H_DIM + hi * 16;
    const _Float16* wbase = WtG + (size_t)(nbase + l16) * H_DIM + hi * 16;

    // prologue: fragments for kb = 0
    v16h a0 = load_a_frag(xrow);
    v16h b0[8];
    #pragma unroll
    for (int j = 0; j < 8; ++j)
      b0[j] = *reinterpret_cast<const v16h*>(wbase + (size_t)j * 16 * H_DIM);

    for (int kb = 0; kb < H_DIM / 32; ++kb) {
      // prefetch kb+1 (last iteration harmlessly re-fetches k=0: no branch)
      const int kn = ((kb + 1) & 31) * 32;
      v16h a1 = load_a_frag(xrow + kn);
      v16h b1[8];
      #pragma unroll
      for (int j = 0; j < 8; ++j)
        b1[j] = *reinterpret_cast<const v16h*>(wbase + (size_t)j * 16 * H_DIM + kn);

      // consume kb
      #pragma unroll
      for (int j = 0; j < 8; ++j)
        acc[j] = __builtin_amdgcn_wmma_f32_16x16x32_f16(
            false, a0, false, b0[j], (short)0, acc[j], false, false);

      a0 = a1;
      #pragma unroll
      for (int j = 0; j < 8; ++j) b0[j] = b1[j];
    }

    #pragma unroll
    for (int j = 0; j < 8; ++j) {
      const int n = nbase + j * 16 + l16;
      #pragma unroll
      for (int r = 0; r < 8; ++r)
        eL[(size_t)(wrow + r + 8 * hi) * E_STRIDE + n] = (_Float16)acc[j][r];
    }
  }
  __syncthreads();

  routing_phase(eL, cvec, bij, dvec, bpart, tid, cap, out);
}

// ---------------------------------------------------------------------------
// Fallback (ws too small): LDS staging of W per k-step (round-1 scheme).
// ---------------------------------------------------------------------------
static constexpr size_t SMEM_LDS =
    (size_t)S_DIM * E_STRIDE * 2 + 128 * WT_STRIDE * 2 +
    H_DIM * 4 + S_DIM * 4 + S_DIM * 4 + 256 * 4;

__global__ __launch_bounds__(256, 1)
void caps_fused_lds(const float* __restrict__ x, const float* __restrict__ W,
                    float* __restrict__ out) {
  extern __shared__ char smem[];
  _Float16* eL    = (_Float16*)smem;
  _Float16* Wt    = (_Float16*)(smem + (size_t)S_DIM * E_STRIDE * 2);
  float*    cvec  = (float*)(smem + (size_t)S_DIM * E_STRIDE * 2 + 128 * WT_STRIDE * 2);
  float*    bij   = cvec + H_DIM;
  float*    dvec  = bij + S_DIM;
  float*    bpart = dvec + S_DIM;

  const int tid  = threadIdx.x;
  const int lane = tid & 31;
  const int wv   = tid >> 5;
  const int l16  = lane & 15;
  const int hi   = lane >> 4;
  const int cap  = blockIdx.x;

  const float* xc = x + (size_t)cap * S_DIM * H_DIM;
  if (tid < S_DIM) bij[tid] = 0.0f;

  const int wrow = wv * 16;
  for (int nc = 0; nc < H_DIM / 128; ++nc) {
    const int nbase = nc * 128;
    v8f acc[8];
    #pragma unroll
    for (int j = 0; j < 8; ++j) {
      v8f z = {0.f, 0.f, 0.f, 0.f, 0.f, 0.f, 0.f, 0.f};
      acc[j] = z;
    }

    for (int kb = 0; kb < H_DIM / 32; ++kb) {
      const int k0 = kb * 32;
      __syncthreads();
      {
        const int kk   = tid >> 3;
        const int col0 = (tid & 7) << 4;
        const float* wp = W + (size_t)(k0 + kk) * H_DIM + nbase + col0;
        #pragma unroll
        for (int i = 0; i < 16; i += 4) {
          float4 f = *reinterpret_cast<const float4*>(wp + i);
          Wt[(col0 + i + 0) * WT_STRIDE + kk] = (_Float16)f.x;
          Wt[(col0 + i + 1) * WT_STRIDE + kk] = (_Float16)f.y;
          Wt[(col0 + i + 2) * WT_STRIDE + kk] = (_Float16)f.z;
          Wt[(col0 + i + 3) * WT_STRIDE + kk] = (_Float16)f.w;
        }
      }
      __syncthreads();

      v16h a = load_a_frag(xc + (size_t)(wrow + l16) * H_DIM + k0 + hi * 16);

      #pragma unroll
      for (int j = 0; j < 8; ++j) {
        const _Float16* bp = &Wt[(j * 16 + l16) * WT_STRIDE + hi * 16];
        v8h bb0 = *(const v8h*)bp;
        v8h bb1 = *(const v8h*)(bp + 8);
        v16h b;
        #pragma unroll
        for (int q = 0; q < 8; ++q) { b[q] = bb0[q]; b[8 + q] = bb1[q]; }
        acc[j] = __builtin_amdgcn_wmma_f32_16x16x32_f16(
            false, a, false, b, (short)0, acc[j], false, false);
      }
    }

    #pragma unroll
    for (int j = 0; j < 8; ++j) {
      const int n = nbase + j * 16 + l16;
      #pragma unroll
      for (int r = 0; r < 8; ++r)
        eL[(size_t)(wrow + r + 8 * hi) * E_STRIDE + n] = (_Float16)acc[j][r];
    }
  }
  __syncthreads();

  routing_phase(eL, cvec, bij, dvec, bpart, tid, cap, out);
}

extern "C" void kernel_launch(void* const* d_in, const int* in_sizes, int n_in,
                              void* d_out, int out_size, void* d_ws, size_t ws_size,
                              hipStream_t stream) {
  const float* x = (const float*)d_in[0];
  const float* W = (const float*)d_in[1];
  float* out = (float*)d_out;

  constexpr size_t WT_BYTES = (size_t)H_DIM * H_DIM * sizeof(_Float16);  // 2 MB
  if (ws_size >= WT_BYTES) {
    _Float16* WtG = (_Float16*)d_ws;
    transpose_w_f16<<<dim3(256), dim3(256), 0, stream>>>(W, WtG);
    caps_fused_fast<<<dim3(C_CAPS), dim3(256), SMEM_FAST, stream>>>(x, WtG, out);
  } else {
    caps_fused_lds<<<dim3(C_CAPS), dim3(256), SMEM_LDS, stream>>>(x, W, out);
  }
}
